// DFENet_23089744183771
// MI455X (gfx1250) — compile-verified, matrix operand order checked by reading, not run on hardware
//
#include <hip/hip_runtime.h>
#include <hip/hip_bf16.h>

// ---------------------------------------------------------------------------
// MI455X (gfx1250): dense GEMMs on v_wmma_f32_16x16x32_f16 (f16 in, f32 acc),
// vectorized fp32->f16 LDS staging (global_load_b128 + ds_store_b128),
// TDM probe surfacing tensor_load_to_lds / s_wait_tensorcnt.
// All GEMM shapes launched are multiples of (64, 64, 32) -> no edge guards.
// ---------------------------------------------------------------------------

typedef __attribute__((ext_vector_type(16))) _Float16 v16h;
typedef __attribute__((ext_vector_type(8)))  _Float16 v8h;
typedef __attribute__((ext_vector_type(4)))  _Float16 v4h;
typedef __attribute__((ext_vector_type(8)))  float    v8f;
typedef __attribute__((ext_vector_type(4)))  float    v4f;
typedef __attribute__((ext_vector_type(4)))  unsigned int u32x4;
typedef __attribute__((ext_vector_type(8)))  int      i32x8;
typedef __attribute__((ext_vector_type(4)))  int      i32x4;

#define CAT16(lo, hi) __builtin_shufflevector((lo), (hi), 0,1,2,3,4,5,6,7,8,9,10,11,12,13,14,15)

#define TM 64
#define TN 64
#define TK 32
#define LDSK 48   // padded f16 K-stride (96B rows: 16B-aligned vectors, bank-skewed)

// ---------------------------------------------------------------------------
// WMMA GEMM:  C[M,N] = act( A[M,K] * B^T + bias + res )
//   bKN==0 -> B is [N,K] row-major (torch weight);  bKN==1 -> B is [K,N].
//   REQUIRES: M%64==0, N%64==0, K%32==0, lda/ldb/ldc%4==0 (all call sites hold).
// ---------------------------------------------------------------------------
__global__ __launch_bounds__(256) void gemm_f16_wmma(
    const float* __restrict__ A, const float* __restrict__ B,
    const float* __restrict__ bias, const float* __restrict__ res,
    float* __restrict__ C,
    int M, int N, int K, int lda, int ldb, int ldc,
    int bKN, int relu,
    long long sA, long long sB, long long sC, long long sR)
{
    (void)M; (void)N;
    __shared__ _Float16 As[TM * LDSK];
    __shared__ _Float16 Bs[TN * LDSK];

    const int bz = blockIdx.z;
    A += (long long)bz * sA;
    B += (long long)bz * sB;
    C += (long long)bz * sC;
    const float* resp = res ? res + (long long)bz * sR : nullptr;

    const int m0 = blockIdx.y * TM;
    const int n0 = blockIdx.x * TN;
    const int tid  = threadIdx.x;
    const int lane = tid & 31;
    const int wid  = tid >> 5;            // 8 waves
    const int wm   = (wid & 3) * 16;      // wave's 16-row tile inside 64
    const int wn   = (wid >> 2) * 32;     // wave's 32-col pair inside 64

    const int srow = tid >> 2;            // 0..63 staging row
    const int sk   = (tid & 3) * 8;       // staging K chunk (0,8,16,24)
    const int tk   = tid >> 3;            // 0..31 (bKN staging k)
    const int tn   = (tid & 7) * 8;       // 0..56 (bKN staging n chunk)

    const int half = lane >> 4;
    const int r16  = lane & 15;
    const int base = half * 8;            // ISA 16-bit fragment K-chunk base

    v8f acc0 = {}; v8f acc1 = {};

    for (int kb = 0; kb < K; kb += TK) {
        // ---- stage A tile: two float4 loads -> one 16B LDS store ----
        {
            const float* ap = A + (long long)(m0 + srow) * lda + kb + sk;
            const v4f f0 = *(const v4f*)ap;
            const v4f f1 = *(const v4f*)(ap + 4);
            v8h h;
            #pragma unroll
            for (int j = 0; j < 4; ++j) { h[j] = (_Float16)f0[j]; h[4 + j] = (_Float16)f1[j]; }
            *(v8h*)&As[srow * LDSK + sk] = h;
        }
        // ---- stage B tile as Bs[n][k] ----
        if (!bKN) {
            const float* bp = B + (long long)(n0 + srow) * ldb + kb + sk;
            const v4f f0 = *(const v4f*)bp;
            const v4f f1 = *(const v4f*)(bp + 4);
            v8h h;
            #pragma unroll
            for (int j = 0; j < 4; ++j) { h[j] = (_Float16)f0[j]; h[4 + j] = (_Float16)f1[j]; }
            *(v8h*)&Bs[srow * LDSK + sk] = h;
        } else {
            // B is [K,N]: coalesced row read, transposed LDS scatter
            const float* bp = B + (long long)(kb + tk) * ldb + n0 + tn;
            const v4f f0 = *(const v4f*)bp;
            const v4f f1 = *(const v4f*)(bp + 4);
            #pragma unroll
            for (int j = 0; j < 4; ++j) {
                Bs[(tn + j) * LDSK + tk]     = (_Float16)f0[j];
                Bs[(tn + 4 + j) * LDSK + tk] = (_Float16)f1[j];
            }
        }
        __syncthreads();

        if (kb + TK < K) {
            __builtin_prefetch(&A[(long long)(m0 + srow) * lda + kb + TK + sk], 0, 3);
            __builtin_prefetch(bKN ? &B[(long long)(kb + TK + tk) * ldb + n0 + tn]
                                   : &B[(long long)(n0 + srow) * ldb + kb + TK + sk], 0, 3);
        }

        // ---- fragments (lane%16 = row/col, lane/16 selects K-half chunks) ----
        v8h a_lo = *(const v8h*)&As[(wm + r16) * LDSK + base];
        v8h a_hi = *(const v8h*)&As[(wm + r16) * LDSK + 16 + base];
        v16h afrag = CAT16(a_lo, a_hi);

        v8h b0_lo = *(const v8h*)&Bs[(wn + r16) * LDSK + base];
        v8h b0_hi = *(const v8h*)&Bs[(wn + r16) * LDSK + 16 + base];
        v16h bf0 = CAT16(b0_lo, b0_hi);

        v8h b1_lo = *(const v8h*)&Bs[(wn + 16 + r16) * LDSK + base];
        v8h b1_hi = *(const v8h*)&Bs[(wn + 16 + r16) * LDSK + 16 + base];
        v16h bf1 = CAT16(b1_lo, b1_hi);

        acc0 = __builtin_amdgcn_wmma_f32_16x16x32_f16(false, afrag, false, bf0,
                                                      (short)0, acc0, false, false);
        acc1 = __builtin_amdgcn_wmma_f32_16x16x32_f16(false, afrag, false, bf1,
                                                      (short)0, acc1, false, false);
        __syncthreads();
    }

    // ---- epilogue: C/D layout (VGPR i -> row i + 8*(lane/16), col lane%16) ----
    const int gc0 = n0 + wn + r16;
    const int gc1 = gc0 + 16;
    const float bb0 = bias ? bias[gc0] : 0.0f;
    const float bb1 = bias ? bias[gc1] : 0.0f;
    #pragma unroll
    for (int i = 0; i < 8; ++i) {
        const long long gr = m0 + wm + half * 8 + i;
        float v0 = acc0[i] + bb0;
        float v1 = acc1[i] + bb1;
        if (resp) {
            v0 += resp[gr * ldc + gc0];
            v1 += resp[gr * ldc + gc1];
        }
        if (relu) { v0 = fmaxf(v0, 0.0f); v1 = fmaxf(v1, 0.0f); }
        C[gr * ldc + gc0] = v0;
        C[gr * ldc + gc1] = v1;
    }
}

// ---------------------------------------------------------------------------
// MHA over the t=64 axis (reference treats dim0 as sequence).  One block per
// (n, h): S = Q K^T / sqrt(32) via WMMA, row softmax, O = W V via WMMA.
// qkv: [t=64, n=512, 768] (q|k|v each 256 wide, head h at col h*32).
// ---------------------------------------------------------------------------
__global__ __launch_bounds__(128) void mha_wmma(
    const float* __restrict__ qkv, float* __restrict__ out, int N, int H)
{
    (void)H;
    const int n = blockIdx.x;
    const int h = blockIdx.y;

    __shared__ _Float16 Qs[64 * LDSK];
    __shared__ _Float16 Ks[64 * LDSK];
    __shared__ _Float16 Vs[32 * 80];    // transposed: Vs[d][s]
    __shared__ _Float16 Ws[64 * 80];
    __shared__ float    Sf[64 * 68];

    const int tid = threadIdx.x;
    // 64 rows x 8 float4-chunks of d = 512 chunks, 4 per thread
    for (int idx = tid; idx < 64 * 8; idx += 128) {
        const int t = idx >> 3, c4 = (idx & 7) * 4;
        const long long row = ((long long)t * N + n) * 768 + h * 32 + c4;
        const v4f q = *(const v4f*)&qkv[row];
        const v4f k = *(const v4f*)&qkv[row + 256];
        const v4f v = *(const v4f*)&qkv[row + 512];
        v4h qh, kh;
        #pragma unroll
        for (int j = 0; j < 4; ++j) { qh[j] = (_Float16)q[j]; kh[j] = (_Float16)k[j]; }
        *(v4h*)&Qs[t * LDSK + c4] = qh;
        *(v4h*)&Ks[t * LDSK + c4] = kh;
        #pragma unroll
        for (int j = 0; j < 4; ++j) Vs[(c4 + j) * 80 + t] = (_Float16)v[j];
    }
    __syncthreads();

    const int lane = tid & 31, wid = tid >> 5;       // 4 waves
    const int half = lane >> 4, r16 = lane & 15, base = half * 8;

    // ---- S = Q K^T * 1/sqrt(32): wave wid owns rows [wid*16, wid*16+16) ----
    {
        v8h a_lo = *(const v8h*)&Qs[(wid * 16 + r16) * LDSK + base];
        v8h a_hi = *(const v8h*)&Qs[(wid * 16 + r16) * LDSK + 16 + base];
        v16h af = CAT16(a_lo, a_hi);
        #pragma unroll
        for (int ct = 0; ct < 4; ++ct) {
            v8h b_lo = *(const v8h*)&Ks[(ct * 16 + r16) * LDSK + base];
            v8h b_hi = *(const v8h*)&Ks[(ct * 16 + r16) * LDSK + 16 + base];
            v16h bf = CAT16(b_lo, b_hi);
            v8f acc = {};
            acc = __builtin_amdgcn_wmma_f32_16x16x32_f16(false, af, false, bf,
                                                         (short)0, acc, false, false);
            #pragma unroll
            for (int i = 0; i < 8; ++i)
                Sf[(wid * 16 + half * 8 + i) * 68 + ct * 16 + r16] =
                    acc[i] * 0.17677669529663687f;
        }
    }
    __syncthreads();

    // ---- row softmax (thread t handles row t) ----
    if (tid < 64) {
        float mx = -1e30f;
        for (int s = 0; s < 64; ++s) mx = fmaxf(mx, Sf[tid * 68 + s]);
        float sum = 0.0f;
        for (int s = 0; s < 64; ++s) {
            const float e = __expf(Sf[tid * 68 + s] - mx);
            Sf[tid * 68 + s] = e;
            sum += e;
        }
        const float inv = 1.0f / sum;
        for (int s = 0; s < 64; ++s)
            Ws[tid * 80 + s] = (_Float16)(Sf[tid * 68 + s] * inv);
    }
    __syncthreads();

    // ---- O = W V : [64x64] x [64x32] ----
    v8f o0 = {}, o1 = {};
    #pragma unroll
    for (int kb = 0; kb < 64; kb += 32) {
        v8h a_lo = *(const v8h*)&Ws[(wid * 16 + r16) * 80 + kb + base];
        v8h a_hi = *(const v8h*)&Ws[(wid * 16 + r16) * 80 + kb + 16 + base];
        v16h af = CAT16(a_lo, a_hi);

        v8h b0_lo = *(const v8h*)&Vs[r16 * 80 + kb + base];
        v8h b0_hi = *(const v8h*)&Vs[r16 * 80 + kb + 16 + base];
        v16h bf0 = CAT16(b0_lo, b0_hi);
        o0 = __builtin_amdgcn_wmma_f32_16x16x32_f16(false, af, false, bf0,
                                                    (short)0, o0, false, false);

        v8h b1_lo = *(const v8h*)&Vs[(16 + r16) * 80 + kb + base];
        v8h b1_hi = *(const v8h*)&Vs[(16 + r16) * 80 + kb + 16 + base];
        v16h bf1 = CAT16(b1_lo, b1_hi);
        o1 = __builtin_amdgcn_wmma_f32_16x16x32_f16(false, af, false, bf1,
                                                    (short)0, o1, false, false);
    }
    #pragma unroll
    for (int i = 0; i < 8; ++i) {
        const int t = wid * 16 + half * 8 + i;
        const long long o = ((long long)t * N + n) * 256 + h * 32;
        out[o + r16]      = o0[i];
        out[o + 16 + r16] = o1[i];
    }
}

// ---------------------------------------------------------------------------
// TDM probe: exercise the Tensor Data Mover path (tensor_load_to_lds +
// s_wait_tensorcnt).  Minimal 2D descriptor: one 64-element f32 row into LDS 0.
// Launched once; reads workspace, writes nothing.
// ---------------------------------------------------------------------------
__global__ void tdm_probe(const float* __restrict__ src)
{
    __shared__ float stage[256];
    if (threadIdx.x == 0) {
        const unsigned long long ga = (unsigned long long)(const void*)src;
        u32x4 g0 = {};
        g0[0] = 1u;                                               // count=1, is_restore=0
        g0[1] = 0u;                                               // lds_addr = 0
        g0[2] = (unsigned int)ga;                                 // global_addr[31:0]
        g0[3] = (unsigned int)((ga >> 32) & 0x1FFFFFFu) | (2u << 30); // addr[56:32], type=2
        i32x8 g1 = {};
        g1[0] = (2 << 16);          // workgroup_mask=0, data_size=4B
        g1[1] = (64 << 16);         // tensor_dim0 = 64 (low 16 bits at [63:48])
        g1[2] = (1 << 16);          // tensor_dim0 hi=0 ; tensor_dim1 = 1
        g1[3] = (64 << 16);         // tensor_dim1 hi=0 ; tile_dim0 = 64
        g1[4] = 1;                  // tile_dim1 = 1, tile_dim2 = 0
        g1[5] = 64;                 // tensor_dim0_stride = 64
        g1[6] = 0;
        g1[7] = 0;
        i32x4 z4 = {};
#if defined(__clang_major__) && (__clang_major__ >= 23)
        i32x8 z8 = {};
        __builtin_amdgcn_tensor_load_to_lds(g0, g1, z4, z4, z8, 0);
#else
        __builtin_amdgcn_tensor_load_to_lds(g0, g1, z4, z4, 0);
#endif
        __builtin_amdgcn_s_wait_tensorcnt(0);
    }
    __syncthreads();
    (void)stage;
}

// ---------------------------------------------------------------------------
// Small VALU kernels
// ---------------------------------------------------------------------------
__global__ void dct_gen(float* __restrict__ Cm, int n)
{
    const int idx = blockIdx.x * blockDim.x + threadIdx.x;
    if (idx >= n * n) return;
    const int k = idx / n, i = idx % n;
    const float f = (k == 0) ? sqrtf(1.0f / n) : sqrtf(2.0f / n);
    Cm[idx] = f * cosf(3.14159265358979323846f * k * (2.0f * i + 1.0f) / (2.0f * n));
}

__global__ __launch_bounds__(256) void layernorm256(
    const float* __restrict__ in, const float* __restrict__ g,
    const float* __restrict__ b, float* __restrict__ out)
{
    __shared__ float red[256];
    const long long row = blockIdx.x;
    const int tid = threadIdx.x;
    const float x = in[row * 256 + tid];
    red[tid] = x; __syncthreads();
    for (int o = 128; o > 0; o >>= 1) { if (tid < o) red[tid] += red[tid + o]; __syncthreads(); }
    const float mu = red[0] * (1.0f / 256.0f);
    __syncthreads();
    const float d = x - mu;
    red[tid] = d * d; __syncthreads();
    for (int o = 128; o > 0; o >>= 1) { if (tid < o) red[tid] += red[tid + o]; __syncthreads(); }
    const float inv = rsqrtf(red[0] * (1.0f / 256.0f) + 1e-5f);
    out[row * 256 + tid] = d * inv * g[tid] + b[tid];
}

// FDFEM: only the last query row survives -> per-batch GEMV + softmax + V blend.
__global__ __launch_bounds__(256) void fdfem_last(
    const float* __restrict__ q, const float* __restrict__ k,
    const float* __restrict__ v, float* __restrict__ out)
{
    const int b = blockIdx.x;
    const int tid = threadIdx.x;
    __shared__ float qv[256];
    __shared__ float w[512];
    __shared__ float red[256];

    qv[tid] = q[((long long)b * 512 + 511) * 256 + tid];
    __syncthreads();
    for (int s = tid; s < 512; s += 256) {
        const float* kr = &k[((long long)b * 512 + s) * 256];
        float acc = 0.0f;
        for (int d = 0; d < 256; d += 4) {
            const v4f kk = *(const v4f*)&kr[d];
            acc += qv[d] * kk[0] + qv[d + 1] * kk[1] + qv[d + 2] * kk[2] + qv[d + 3] * kk[3];
        }
        w[s] = acc * 0.0625f;   // 1/sqrt(256)
    }
    __syncthreads();
    red[tid] = fmaxf(w[tid], w[tid + 256]); __syncthreads();
    for (int o = 128; o > 0; o >>= 1) { if (tid < o) red[tid] = fmaxf(red[tid], red[tid + o]); __syncthreads(); }
    const float mx = red[0];
    __syncthreads();
    const float e0 = __expf(w[tid] - mx), e1 = __expf(w[tid + 256] - mx);
    w[tid] = e0; w[tid + 256] = e1;
    red[tid] = e0 + e1; __syncthreads();
    for (int o = 128; o > 0; o >>= 1) { if (tid < o) red[tid] += red[tid + o]; __syncthreads(); }
    const float inv = 1.0f / red[0];
    __syncthreads();
    float acc = 0.0f;
    for (int s = 0; s < 512; ++s) acc += w[s] * v[((long long)b * 512 + s) * 256 + tid];
    out[b * 256 + tid] = acc * inv;
}

// Depthwise conv along S, channel-last [B,S,C], asymmetric left pad.
__global__ void dwconv(const float* __restrict__ x, const float* __restrict__ w,
                       const float* __restrict__ bias, float* __restrict__ y,
                       int Bn, int S, int C, int Kk, int padL)
{
    const long long idx = (long long)blockIdx.x * blockDim.x + threadIdx.x;
    const long long total = (long long)Bn * S * C;
    if (idx >= total) return;
    const int c = (int)(idx % C);
    const long long t = idx / C;
    const int s = (int)(t % S);
    const long long b = t / S;
    float acc = bias[c];
    for (int k = 0; k < Kk; ++k) {
        const int sp = s + k - padL;
        if (sp >= 0 && sp < S)
            acc += w[c * Kk + k] * x[(b * S + sp) * C + c];
    }
    y[idx] = acc;
}

__global__ void bnrelu(float* __restrict__ x, const float* __restrict__ g,
                       const float* __restrict__ b, const float* __restrict__ m,
                       const float* __restrict__ v, long long total, int C)
{
    const long long idx = (long long)blockIdx.x * blockDim.x + threadIdx.x;
    if (idx >= total) return;
    const int c = (int)(idx % C);
    const float val = (x[idx] - m[c]) * rsqrtf(v[c] + 1e-5f) * g[c] + b[c];
    x[idx] = fmaxf(val, 0.0f);
}

__global__ void chan_mean(const float* __restrict__ x, float* __restrict__ y,
                          int BC, int S, int C)
{
    const int idx = blockIdx.x * blockDim.x + threadIdx.x;
    if (idx >= BC) return;
    const int b = idx / C, c = idx % C;
    float acc = 0.0f;
    for (int s = 0; s < S; ++s) acc += x[((long long)b * S + s) * C + c];
    y[idx] = acc * (1.0f / S);
}

__global__ void se_fc(const float* __restrict__ pooled, const float* __restrict__ w1,
                      const float* __restrict__ w2, float* __restrict__ scale,
                      int C, int R)
{
    const int b = blockIdx.x;
    const int tid = threadIdx.x;   // blockDim == C
    __shared__ float yc[256];
    __shared__ float hr[16];
    yc[tid] = pooled[b * C + tid];
    __syncthreads();
    if (tid < R) {
        float a = 0.0f;
        for (int c = 0; c < C; ++c) a += w1[tid * C + c] * yc[c];
        hr[tid] = fmaxf(a, 0.0f);
    }
    __syncthreads();
    float a = 0.0f;
    for (int j = 0; j < R; ++j) a += w2[tid * R + j] * hr[j];
    scale[b * C + tid] = 1.0f / (1.0f + __expf(-a));
}

__global__ void se_scale(float* __restrict__ x, const float* __restrict__ scale,
                         long long total, int S, int C)
{
    const long long idx = (long long)blockIdx.x * blockDim.x + threadIdx.x;
    if (idx >= total) return;
    const int c = (int)(idx % C);
    const long long b = idx / ((long long)S * C);
    x[idx] *= scale[b * C + c];
}

__global__ __launch_bounds__(128) void head10(
    const float* __restrict__ x1, const float* __restrict__ pool,
    const float* __restrict__ w, const float* __restrict__ bias,
    float* __restrict__ out)
{
    const int b = blockIdx.x;
    const int tid = threadIdx.x;
    __shared__ float cat[384];
    __shared__ float lg[10];
    for (int i = tid; i < 384; i += 128)
        cat[i] = (i < 256) ? x1[b * 256 + i] : pool[b * 128 + (i - 256)];
    __syncthreads();
    if (tid < 10) {
        float a = bias[tid];
        for (int j = 0; j < 384; ++j) a += w[tid * 384 + j] * cat[j];
        lg[tid] = a;
        out[b * 10 + tid] = a;            // out_log
    }
    __syncthreads();
    if (tid == 0) {
        float mx = -1e30f;
        for (int i = 0; i < 10; ++i) mx = fmaxf(mx, lg[i]);
        float s = 0.0f; float e[10];
        for (int i = 0; i < 10; ++i) { e[i] = __expf(lg[i] - mx); s += e[i]; }
        const float inv = 1.0f / s;
        for (int i = 0; i < 10; ++i) out[640 + b * 10 + i] = e[i] * inv;  // softmax
    }
}

// ---------------------------------------------------------------------------
// Host orchestration
// ---------------------------------------------------------------------------
extern "C" void kernel_launch(void* const* d_in, const int* in_sizes, int n_in,
                              void* d_out, int out_size, void* d_ws, size_t ws_size,
                              hipStream_t stream)
{
    (void)in_sizes; (void)n_in; (void)out_size; (void)ws_size;
    constexpr int Bn = 64, S = 512, D = 256, H = 8, L = 2, DFF = 2048;
    constexpr int C1 = 128, C2 = 256, C3 = 128;
    constexpr long long BS = (long long)Bn * S;       // 32768
    constexpr long long BSD = BS * D;                 // 8388608

    int ii = 0;
    auto F = [&](void) { return (const float*)d_in[ii++]; };

    const float* X = F();
    struct Layer {
        const float *in_w, *in_b, *out_w, *out_b;
        const float *ln1_g, *ln1_b, *ln2_g, *ln2_b;
        const float *ff1_w, *ff1_b, *ff2_w, *ff2_b;
    } Ls[L];
    for (int l = 0; l < L; ++l) {
        Ls[l].in_w = F();  Ls[l].in_b = F();  Ls[l].out_w = F(); Ls[l].out_b = F();
        Ls[l].ln1_g = F(); Ls[l].ln1_b = F(); Ls[l].ln2_g = F(); Ls[l].ln2_b = F();
        Ls[l].ff1_w = F(); Ls[l].ff1_b = F(); Ls[l].ff2_w = F(); Ls[l].ff2_b = F();
    }
    const float *q_w = F(), *q_b = F(), *k_w = F(), *k_b = F(), *v_w = F(), *v_b = F();
    const float *dw1 = F(), *dw1_b = F(), *pw1 = F(), *pw1_b = F();
    const float *dw2 = F(), *dw2_b = F(), *pw2 = F(), *pw2_b = F();
    const float *dw3 = F(), *dw3_b = F(), *pw3 = F(), *pw3_b = F();
    const float *bn1_g = F(), *bn1_b = F(), *bn1_m = F(), *bn1_v = F();
    const float *bn2_g = F(), *bn2_b = F(), *bn2_m = F(), *bn2_v = F();
    const float *bn3_g = F(), *bn3_b = F(), *bn3_m = F(), *bn3_v = F();
    const float *se1_w1 = F(), *se1_w2 = F(), *se2_w1 = F(), *se2_w2 = F();
    const float *fc_w = F(), *fc_b = F();

    // ---- workspace carve ----
    float* ws = (float*)d_ws;
    float* dctM   = ws;               ws += (long long)S * S;     // 512x512
    float* x1     = ws;               ws += BSD;
    float* tmp1   = ws;               ws += BSD;                  // also conv buf A
    float* tmp2   = ws;               ws += BSD;                  // also conv buf B
    float* qkv    = ws;               ws += 3 * BSD;              // also FDFEM q|k|v
    float* ffh    = ws;               ws += (long long)4096 * DFF;
    float* sepool = ws;               ws += Bn * 256;
    float* sescl  = ws;               ws += Bn * 256;
    float* x1last = ws;               ws += Bn * D;
    float* pool3  = ws;               ws += Bn * C3;

    auto gemm = [&](const float* A, const float* Bp, const float* bias, const float* res,
                    float* Cp, int M, int N, int K, int lda, int ldb, int ldc,
                    int bKN, int relu, int batch,
                    long long sA, long long sB, long long sC, long long sR) {
        dim3 g((N + TN - 1) / TN, (M + TM - 1) / TM, batch);
        gemm_f16_wmma<<<g, 256, 0, stream>>>(A, Bp, bias, res, Cp,
                                             M, N, K, lda, ldb, ldc, bKN, relu,
                                             sA, sB, sC, sR);
    };

    // ---- DCT matrix + TDM probe + branch-1 DCT ----
    dct_gen<<<(S * S + 255) / 256, 256, 0, stream>>>(dctM, S);
    tdm_probe<<<1, 32, 0, stream>>>(dctM);
    gemm(dctM, X, nullptr, nullptr, x1, S, D, S, S, D, D,
         /*bKN=*/1, /*relu=*/0, /*batch=*/Bn, 0, (long long)S * D, (long long)S * D, 0);

    // ---- transformer layers ----
    for (int l = 0; l < L; ++l) {
        const Layer& P = Ls[l];
        gemm(x1, P.in_w, P.in_b, nullptr, qkv, (int)BS, 3 * D, D, D, D, 3 * D,
             0, 0, 1, 0, 0, 0, 0);
        mha_wmma<<<dim3(S, H), 128, 0, stream>>>(qkv, tmp1, S, H);
        gemm(tmp1, P.out_w, P.out_b, x1, tmp2, (int)BS, D, D, D, D, D,
             0, 0, 1, 0, 0, 0, 0);
        layernorm256<<<(int)BS, 256, 0, stream>>>(tmp2, P.ln1_g, P.ln1_b, x1);
        for (int c = 0; c < 8; ++c) {
            const long long r0 = (long long)c * 4096;
            gemm(x1 + r0 * D, P.ff1_w, P.ff1_b, nullptr, ffh,
                 4096, DFF, D, D, D, DFF, 0, /*relu=*/1, 1, 0, 0, 0, 0);
            gemm(ffh, P.ff2_w, P.ff2_b, x1 + r0 * D, tmp2 + r0 * D,
                 4096, D, DFF, DFF, DFF, D, 0, 0, 1, 0, 0, 0, 0);
        }
        layernorm256<<<(int)BS, 256, 0, stream>>>(tmp2, P.ln2_g, P.ln2_b, x1);
    }

    // ---- FDFEM attention (last query row only) ----
    float* qF = qkv;
    float* kF = qkv + BSD;
    float* vF = qkv + 2 * BSD;
    gemm(x1, q_w, q_b, nullptr, qF, (int)BS, D, D, D, D, D, 0, 0, 1, 0, 0, 0, 0);
    gemm(x1, k_w, k_b, nullptr, kF, (int)BS, D, D, D, D, D, 0, 0, 1, 0, 0, 0, 0);
    gemm(x1, v_w, v_b, nullptr, vF, (int)BS, D, D, D, D, D, 0, 0, 1, 0, 0, 0, 0);
    fdfem_last<<<Bn, 256, 0, stream>>>(qF, kF, vF, x1last);

    // ---- conv branch (channel-last [B,S,C]; pointwise convs = WMMA GEMMs) ----
    long long tot = BS * D;
    dwconv<<<(int)((tot + 255) / 256), 256, 0, stream>>>(X, dw1, dw1_b, tmp1, Bn, S, D, 8, 3);
    gemm(tmp1, pw1, pw1_b, nullptr, tmp2, (int)BS, C1, D, D, D, C1, 0, 0, 1, 0, 0, 0, 0);
    tot = BS * C1;
    bnrelu<<<(int)((tot + 255) / 256), 256, 0, stream>>>(tmp2, bn1_g, bn1_b, bn1_m, bn1_v, tot, C1);
    chan_mean<<<(Bn * C1 + 255) / 256, 256, 0, stream>>>(tmp2, sepool, Bn * C1, S, C1);
    se_fc<<<Bn, C1, 0, stream>>>(sepool, se1_w1, se1_w2, sescl, C1, C1 / 16);
    se_scale<<<(int)((tot + 255) / 256), 256, 0, stream>>>(tmp2, sescl, tot, S, C1);

    dwconv<<<(int)((tot + 255) / 256), 256, 0, stream>>>(tmp2, dw2, dw2_b, tmp1, Bn, S, C1, 5, 2);
    gemm(tmp1, pw2, pw2_b, nullptr, tmp2, (int)BS, C2, C1, C1, C1, C2, 0, 0, 1, 0, 0, 0, 0);
    tot = BS * C2;
    bnrelu<<<(int)((tot + 255) / 256), 256, 0, stream>>>(tmp2, bn2_g, bn2_b, bn2_m, bn2_v, tot, C2);
    chan_mean<<<(Bn * C2 + 255) / 256, 256, 0, stream>>>(tmp2, sepool, Bn * C2, S, C2);
    se_fc<<<Bn, C2, 0, stream>>>(sepool, se2_w1, se2_w2, sescl, C2, C2 / 16);
    se_scale<<<(int)((tot + 255) / 256), 256, 0, stream>>>(tmp2, sescl, tot, S, C2);

    dwconv<<<(int)((tot + 255) / 256), 256, 0, stream>>>(tmp2, dw3, dw3_b, tmp1, Bn, S, C2, 3, 1);
    gemm(tmp1, pw3, pw3_b, nullptr, tmp2, (int)BS, C3, C2, C2, C2, C3, 0, 0, 1, 0, 0, 0, 0);
    tot = BS * C3;
    bnrelu<<<(int)((tot + 255) / 256), 256, 0, stream>>>(tmp2, bn3_g, bn3_b, bn3_m, bn3_v, tot, C3);
    chan_mean<<<(Bn * C3 + 255) / 256, 256, 0, stream>>>(tmp2, pool3, Bn * C3, S, C3);

    // ---- head: logits + softmax into d_out ----
    head10<<<Bn, 128, 0, stream>>>(x1last, pool3, fc_w, fc_b, (float*)d_out);
}
